// GATNet_85538568667546
// MI455X (gfx1250) — matrix operand order repaired and checked by reading, not compiled, not run on hardware
//
#include <hip/hip_runtime.h>
#include <hip/hip_bf16.h>
#include <cstdint>

typedef __attribute__((ext_vector_type(16))) _Float16 v16h;
typedef __attribute__((ext_vector_type(8)))  _Float16 v8h;
typedef __attribute__((ext_vector_type(8)))  float    v8f;
typedef int gcc_v4i __attribute__((vector_size(16)));   // matches builtin param type

#define GAT_N  50000
#define GAT_E  800000
#define GAT_D  128

#if __has_builtin(__builtin_amdgcn_global_load_async_to_lds_b128)
#define GFX1250_ASYNC 1
#else
#define GFX1250_ASYNC 0
#endif

#define AS1 __attribute__((address_space(1)))
#define AS3 __attribute__((address_space(3)))

static __device__ __forceinline__ void async_copy16(const void* g, void* l) {
#if GFX1250_ASYNC
    __builtin_amdgcn_global_load_async_to_lds_b128(
        (AS1 gcc_v4i*)(uintptr_t)g, (AS3 gcc_v4i*)(uintptr_t)l, /*offset=*/0, /*cpol=*/0);
#else
    *(v8h*)l = *(const v8h*)g;
#endif
}

static __device__ __forceinline__ void async_wait0() {
#if GFX1250_ASYNC && __has_builtin(__builtin_amdgcn_s_wait_asynccnt)
    __builtin_amdgcn_s_wait_asynccnt(0);
#endif
}

static __device__ __forceinline__ float lrelu02(float x) { return x > 0.f ? x : 0.2f * x; }

static __device__ __forceinline__ void atomicMaxFloat(float* addr, float value) {
    // monotonic bit-pattern trick; valid with -1e30f init
    if (value >= 0.f) atomicMax((int*)addr, __float_as_int(value));
    else              atomicMin((unsigned int*)addr, __float_as_uint(value));
}

// ---------------------------------------------------------------------------
// WMMA GEMM: C[M,Nc] = A[M,K] * B[K,Nc]  (+bias, optional relu)
// A, B are f16; accumulation f32 via v_wmma_f32_16x16x32_f16.
// Block = 5 waves sharing one 16-wide N-tile; each wave owns one 16-row
// M-tile. B tile (32x16 f16 = 1KB) is async-staged to LDS, double buffered.
// A fragment = two global_load_b128 per lane per K-step (layout-exact).
// M%16==0, K%32==0, Nc%16==0 for all call sites.
// ---------------------------------------------------------------------------
#define GEMM_WAVES 5

__global__ void gat_gemm_wmma(const _Float16* __restrict__ A, const _Float16* __restrict__ B,
                              const float* __restrict__ bias, float* __restrict__ C,
                              int M, int Nc, int K, int fuse_relu) {
    __shared__ __align__(16) _Float16 sB[2][32 * 16];

    const int lane   = threadIdx.x & 31;
    const int wave   = threadIdx.x >> 5;
    const int tilesM = M >> 4;
    const int tn     = blockIdx.x << 4;                 // N-tile for whole block
    const int tmIdx  = blockIdx.y * GEMM_WAVES + wave;  // wave-uniform
    const int tm     = (tmIdx < tilesM ? tmIdx : 0) << 4;

    const int t    = threadIdx.x;
    const int srow = t >> 1;          // 0..31  (staging threads t<64)
    const int scol = (t & 1) << 3;    // 0 or 8 halfs

    if (t < 64) async_copy16(B + (size_t)srow * Nc + tn + scol, &sB[0][srow * 16 + scol]);

    const int half = lane >> 4;                     // 0: lanes 0-15, 1: lanes 16-31
    const _Float16* arow = A + (size_t)(tm + (lane & 15)) * K;

    v8f acc = {};
    int buf = 0;
    for (int k0 = 0; k0 < K; k0 += 32, buf ^= 1) {
        async_wait0();        // stagers: own async copy landed in LDS
        __syncthreads();      // publish tile to all 5 waves
        if (k0 + 32 < K && t < 64)
            async_copy16(B + (size_t)(k0 + 32 + srow) * Nc + tn + scol,
                         &sB[buf ^ 1][srow * 16 + scol]);

        // A fragment: a[0..7] = K(half*8 .. +7), a[8..15] = K(16+half*8 .. +7)
        const v8h alo = *(const v8h*)(arow + k0 + half * 8);
        const v8h ahi = *(const v8h*)(arow + k0 + 16 + half * 8);
        // B fragment: column N = lane&15, b[0..7] = K(half*8 .. +7), b[8..15] = K(16+half*8 .. +7)
        v16h b;
        for (int s = 0; s < 8; ++s) {
            b[s]     = sB[buf][(half * 8 + s) * 16 + (lane & 15)];
            b[s + 8] = sB[buf][(16 + half * 8 + s) * 16 + (lane & 15)];
        }
        const v16h a = __builtin_shufflevector(alo, ahi, 0,1,2,3,4,5,6,7,8,9,10,11,12,13,14,15);

        acc = __builtin_amdgcn_wmma_f32_16x16x32_f16(
            /*neg_a=*/false, a, /*neg_b=*/false, b,
            /*c_mod=*/(short)0, acc, /*reuse_a=*/false, /*reuse_b=*/false);
    }

    if (tmIdx < tilesM) {
        const int col = tn + (lane & 15);
        const float bv = bias ? bias[col] : 0.f;
#pragma unroll
        for (int r = 0; r < 8; ++r) {   // VGPR r holds row r (+8 for upper half-lanes)
            float v = acc[r] + bv;
            if (fuse_relu) v = fmaxf(v, 0.f);
            C[(size_t)(tm + r + (half << 3)) * Nc + col] = v;
        }
    }
}

// ---------------------------------------------------------------------------
__global__ void gat_f32_to_f16(const float* __restrict__ src, _Float16* __restrict__ dst, int n) {
    const int i = blockIdx.x * blockDim.x + threadIdx.x;
    if (i < n) dst[i] = (_Float16)src[i];
}

// Per-node attention logits: asrc[n,h] = <H[n,h,:], att_src[h,:]>, same for dst
__global__ void gat_alpha(const float* __restrict__ H, const float* __restrict__ att_src,
                          const float* __restrict__ att_dst, float* __restrict__ asrc,
                          float* __restrict__ adst, int n_nodes, int heads) {
    const int i = blockIdx.x * blockDim.x + threadIdx.x;
    if (i >= n_nodes * heads) return;
    const int h = i % heads;
    const float* hp = H + (size_t)i * GAT_D;          // [n, heads, D] contiguous
    const float* as = att_src + h * GAT_D;
    const float* ad = att_dst + h * GAT_D;
    float s = 0.f, d = 0.f;
#pragma unroll 4
    for (int k = 0; k < GAT_D; ++k) { const float v = hp[k]; s += v * as[k]; d += v * ad[k]; }
    asrc[i] = s;
    adst[i] = d;
}

__global__ void gat_fill(float* __restrict__ p, float v, int n) {
    const int i = blockIdx.x * blockDim.x + threadIdx.x;
    if (i < n) p[i] = v;
}

// edge id -> (src,dst); e >= E means self-loop (e - E)
static __device__ __forceinline__ void edge_sd(const int* __restrict__ ei, int e, int& s, int& d) {
    if (e < GAT_E) { s = ei[e]; d = ei[GAT_E + e]; }
    else           { s = d = e - GAT_E; }
}

__global__ void gat_edge_max(const int* __restrict__ ei, const float* __restrict__ asrc,
                             const float* __restrict__ adst, float* __restrict__ nmax, int heads) {
    const int ET = GAT_E + GAT_N;
    const int i = blockIdx.x * blockDim.x + threadIdx.x;
    if (i >= ET * heads) return;
    const int e = i / heads, h = i % heads;
    int s, d; edge_sd(ei, e, s, d);
    const float v = lrelu02(asrc[s * heads + h] + adst[d * heads + h]);
    atomicMaxFloat(&nmax[d * heads + h], v);
}

__global__ void gat_edge_exp(const int* __restrict__ ei, const float* __restrict__ asrc,
                             const float* __restrict__ adst, const float* __restrict__ nmax,
                             float* __restrict__ ex, float* __restrict__ denom, int heads) {
    const int ET = GAT_E + GAT_N;
    const int i = blockIdx.x * blockDim.x + threadIdx.x;
    if (i >= ET * heads) return;
    const int e = i / heads, h = i % heads;
    int s, d; edge_sd(ei, e, s, d);
    const float v  = lrelu02(asrc[s * heads + h] + adst[d * heads + h]);
    const float xv = __expf(v - nmax[d * heads + h]);
    ex[i] = xv;
    atomicAdd(&denom[d * heads + h], xv);
}

// one thread per (edge, head, 4-float feature chunk)
__global__ void gat_edge_agg(const int* __restrict__ ei, const float* __restrict__ H,
                             const float* __restrict__ ex, const float* __restrict__ denom,
                             float* __restrict__ agg, int heads) {
    const int CH = GAT_D / 4;  // 32 chunks
    const int ET = GAT_E + GAT_N;
    const long long i = (long long)blockIdx.x * blockDim.x + threadIdx.x;
    const long long total = (long long)ET * heads * CH;
    if (i >= total) return;
    const int c = (int)(i % CH);
    const long long rest = i / CH;
    const int h = (int)(rest % heads);
    const int e = (int)(rest / heads);
    int s, d; edge_sd(ei, e, s, d);
    const float alpha = ex[(size_t)e * heads + h] / fmaxf(denom[d * heads + h], 1e-16f);
    const float4 hv = *(const float4*)(H + ((size_t)s * heads + h) * GAT_D + c * 4);
    float* ap = agg + ((size_t)d * heads + h) * GAT_D + c * 4;
    atomicAdd(ap + 0, hv.x * alpha);
    atomicAdd(ap + 1, hv.y * alpha);
    atomicAdd(ap + 2, hv.z * alpha);
    atomicAdd(ap + 3, hv.w * alpha);
}

// elu(agg + bias) -> f16 activations for the next WMMA GEMM
__global__ void gat_elu_bias_f16(const float* __restrict__ agg, const float* __restrict__ bias,
                                 _Float16* __restrict__ outh, int n_nodes, int C_) {
    const int i = blockIdx.x * blockDim.x + threadIdx.x;
    if (i >= n_nodes * C_) return;
    const float v = agg[i] + bias[i % C_];
    outh[i] = (_Float16)(v > 0.f ? v : (__expf(v) - 1.f));
}

// final head: out[n,c] = relu(<h3[n,:], Wm2[:,c]> + bm2[c]), C=8
__global__ void gat_mlp_out(const float* __restrict__ H, const float* __restrict__ Wm2,
                            const float* __restrict__ bm2, float* __restrict__ out, int total) {
    const int i = blockIdx.x * blockDim.x + threadIdx.x;
    if (i >= total) return;
    const int node = i >> 3, c = i & 7;
    const float* hp = H + (size_t)node * GAT_D;
    float acc = bm2[c];
#pragma unroll 4
    for (int k = 0; k < GAT_D; ++k) acc += hp[k] * Wm2[k * 8 + c];
    out[i] = fmaxf(acc, 0.f);
}

// ---------------------------------------------------------------------------
extern "C" void kernel_launch(void* const* d_in, const int* in_sizes, int n_in,
                              void* d_out, int out_size, void* d_ws, size_t ws_size,
                              hipStream_t stream) {
    const float* x        = (const float*)d_in[0];
    const int*   ei       = (const int*)d_in[1];
    const float* W1       = (const float*)d_in[2];
    const float* att1_src = (const float*)d_in[3];
    const float* att1_dst = (const float*)d_in[4];
    const float* b1       = (const float*)d_in[5];
    const float* W2       = (const float*)d_in[6];
    const float* att2_src = (const float*)d_in[7];
    const float* att2_dst = (const float*)d_in[8];
    const float* b2       = (const float*)d_in[9];
    const float* Wm1      = (const float*)d_in[10];
    const float* bm1      = (const float*)d_in[11];
    const float* Wm2      = (const float*)d_in[12];
    const float* bm2      = (const float*)d_in[13];
    float* out = (float*)d_out;

    const int N = GAT_N, ET = GAT_E + GAT_N;

    // ---- workspace layout: f32 region, then f16 region -----------------------
    float* fb0   = (float*)d_ws;              // N*256 : h1; later h2 in [0,N*128)
    float* fb1   = fb0 + (size_t)N * 256;     // N*256 : agg1; later agg2 then h3 in [0,N*128)
    float* asrc1 = fb1 + (size_t)N * 256;     // N*2
    float* adst1 = asrc1 + (size_t)N * 2;
    float* nmax1 = adst1 + (size_t)N * 2;
    float* den1  = nmax1 + (size_t)N * 2;
    float* asrc2 = den1 + (size_t)N * 2;      // N each
    float* adst2 = asrc2 + (size_t)N;
    float* nmax2 = adst2 + (size_t)N;
    float* den2  = nmax2 + (size_t)N;
    float* ex1   = den2 + (size_t)N;          // ET*2
    float* ex2   = ex1 + (size_t)ET * 2;      // ET
    uintptr_t hp16 = ((uintptr_t)(ex2 + ET) + 15) & ~(uintptr_t)15;
    _Float16* hb   = (_Float16*)hp16;         // N*256 halfs (xh / x1h / x2h, reused)
    _Float16* wh1  = hb + (size_t)N * 256;    // 128*256
    _Float16* wh2  = wh1 + 128 * 256;         // 256*128
    _Float16* whm1 = wh2 + 256 * 128;         // 128*128

    auto cdiv = [](long long a, long long b) { return (int)((a + b - 1) / b); };
    const int BT = 256;
    const int gemmGridY = cdiv(N / 16, GEMM_WAVES);   // 3125/5 = 625 exact

    // ---- one-time (per launch) f16 conversions -------------------------------
    gat_f32_to_f16<<<cdiv((long long)N * 128, BT), BT, 0, stream>>>(x, hb, N * 128);
    gat_f32_to_f16<<<cdiv(128 * 256, BT), BT, 0, stream>>>(W1, wh1, 128 * 256);
    gat_f32_to_f16<<<cdiv(256 * 128, BT), BT, 0, stream>>>(W2, wh2, 256 * 128);
    gat_f32_to_f16<<<cdiv(128 * 128, BT), BT, 0, stream>>>(Wm1, whm1, 128 * 128);

    // ---------------- Layer 1: GATConv(128 -> 128, heads=2, concat) -----------
    {
        gat_gemm_wmma<<<dim3(256 / 16, gemmGridY), 32 * GEMM_WAVES, 0, stream>>>(
            hb, wh1, nullptr, fb0, N, 256, 128, 0);

        gat_alpha<<<cdiv((long long)N * 2, BT), BT, 0, stream>>>(fb0, att1_src, att1_dst,
                                                                 asrc1, adst1, N, 2);
        gat_fill<<<cdiv((long long)N * 2, BT), BT, 0, stream>>>(nmax1, -1e30f, N * 2);
        gat_fill<<<cdiv((long long)N * 2, BT), BT, 0, stream>>>(den1, 0.f, N * 2);
        gat_fill<<<cdiv((long long)N * 256, BT), BT, 0, stream>>>(fb1, 0.f, N * 256);

        gat_edge_max<<<cdiv((long long)ET * 2, BT), BT, 0, stream>>>(ei, asrc1, adst1, nmax1, 2);
        gat_edge_exp<<<cdiv((long long)ET * 2, BT), BT, 0, stream>>>(ei, asrc1, adst1, nmax1,
                                                                     ex1, den1, 2);
        gat_edge_agg<<<cdiv((long long)ET * 2 * 32, BT), BT, 0, stream>>>(ei, fb0, ex1, den1,
                                                                          fb1, 2);
        gat_elu_bias_f16<<<cdiv((long long)N * 256, BT), BT, 0, stream>>>(fb1, b1, hb, N, 256);
    }

    // ---------------- Layer 2: GATConv(256 -> 128, heads=1) -------------------
    float* h2   = fb0;   // N*128 (overwrites dead h1)
    float* agg2 = fb1;   // N*128 (overwrites dead agg1)
    {
        gat_gemm_wmma<<<dim3(128 / 16, gemmGridY), 32 * GEMM_WAVES, 0, stream>>>(
            hb, wh2, nullptr, h2, N, 128, 256, 0);

        gat_alpha<<<cdiv((long long)N, BT), BT, 0, stream>>>(h2, att2_src, att2_dst,
                                                             asrc2, adst2, N, 1);
        gat_fill<<<cdiv((long long)N, BT), BT, 0, stream>>>(nmax2, -1e30f, N);
        gat_fill<<<cdiv((long long)N, BT), BT, 0, stream>>>(den2, 0.f, N);
        gat_fill<<<cdiv((long long)N * 128, BT), BT, 0, stream>>>(agg2, 0.f, N * 128);

        gat_edge_max<<<cdiv((long long)ET, BT), BT, 0, stream>>>(ei, asrc2, adst2, nmax2, 1);
        gat_edge_exp<<<cdiv((long long)ET, BT), BT, 0, stream>>>(ei, asrc2, adst2, nmax2,
                                                                 ex2, den2, 1);
        gat_edge_agg<<<cdiv((long long)ET * 32, BT), BT, 0, stream>>>(ei, h2, ex2, den2, agg2, 1);
        gat_elu_bias_f16<<<cdiv((long long)N * 128, BT), BT, 0, stream>>>(agg2, b2, hb, N, 128);
        // x2 (f16) now lives in hb
    }

    // ---------------- MLP head -------------------------------------------------
    float* h3 = fb1;     // N*128 (overwrites dead agg2)
    {
        gat_gemm_wmma<<<dim3(128 / 16, gemmGridY), 32 * GEMM_WAVES, 0, stream>>>(
            hb, whm1, bm1, h3, N, 128, 128, 1);
        gat_mlp_out<<<cdiv((long long)N * 8, BT), BT, 0, stream>>>(h3, Wm2, bm2, out, N * 8);
    }
    (void)in_sizes; (void)n_in; (void)out_size; (void)ws_size;
}
